// GatedDIP_stem_gate_12764642804285
// MI455X (gfx1250) — matrix operand order repaired and checked by reading, not compiled, not run on hardware
//
#include <hip/hip_runtime.h>
#include <hip/hip_bf16.h>
#include <math.h>

// ---------------------------------------------------------------------------
// Types for CDNA5 WMMA
// ---------------------------------------------------------------------------
typedef _Float16 v16h __attribute__((ext_vector_type(16)));
typedef _Float16 v8h  __attribute__((ext_vector_type(8)));
typedef float    v8f  __attribute__((ext_vector_type(8)));

// ---------------------------------------------------------------------------
// Small device helpers
// ---------------------------------------------------------------------------
__device__ __forceinline__ float trf(float x, float l, float r) {
    return (tanhf(x) * 0.5f + 0.5f) * (r - l) + l;
}
__device__ __forceinline__ unsigned fkey(float f) {
    unsigned u = __float_as_uint(f);
    return (u & 0x80000000u) ? ~u : (u | 0x80000000u);
}
__device__ __forceinline__ float funkey(unsigned k) {
    unsigned u = (k & 0x80000000u) ? (k & 0x7FFFFFFFu) : ~k;
    return __uint_as_float(u);
}

// ---------------------------------------------------------------------------
// Weight preconversion f32 -> f16 (done once per launch; weights then live in
// L2 as f16, halving the per-block refetch traffic of the conv kernels).
// ---------------------------------------------------------------------------
__global__ __launch_bounds__(256) void convert_w_kernel(
    const float* __restrict__ w, _Float16* __restrict__ wh, int n)
{
    int i = blockIdx.x * 256 + threadIdx.x;
    if (i < n) wh[i] = (_Float16)w[i];
}

// ---------------------------------------------------------------------------
// Generic implicit-GEMM 3x3 conv using WMMA f16.
// GEMM view per batch (blockIdx.z = b):  M = Cout, N = Ho*Wo, K = Cin*9.
// Block = 256 threads = 8 wave32s. Block tile: 64(M) x 64(N), K-step 32.
// Wave (mi,ni): mi in 0..3 -> 16-row M slice, ni in 0..1 -> 32-col N slice
// (two 16x16x32 WMMAs per wave per K-step).
//
// Optional read-side transform on input activations: v = relu(v*sc[c]+sh[c])
// (fused BatchNorm+ReLU of the *previous* layer).
// Epilogue modes: 0: out = acc+bias
//                 1: out = gate[b]*(acc+bias)          (overwrite)
//                 2: out += gate[b]*(acc+bias)         (stream-ordered RMW)
// ---------------------------------------------------------------------------
template <typename TIn, typename TOut>
__global__ __launch_bounds__(256) void conv_wmma_kernel(
    const TIn* __restrict__ in, const _Float16* __restrict__ wf,
    const float* __restrict__ bias, TOut* __restrict__ out,
    const float* __restrict__ insc, const float* __restrict__ insh,
    const float* __restrict__ gate, int mode,
    int Cin, int Hin, int Win, int Cout, int Ho, int Wo,
    int stride, int pad)
{
    // LDS tiles. A: [64 rows(M)][pitch 40] f16 (80B rows, 16B-aligned octets)
    //            B: [64 rows(N)][pitch 48] f16 (96B rows, 16B-aligned octets)
    __shared__ _Float16 As[64 * 40];
    __shared__ _Float16 Bs[64 * 48];

    const int tid  = threadIdx.x;
    const int lane = tid & 31;
    const int wave = tid >> 5;
    const int mi   = wave >> 1;   // 0..3
    const int ni   = wave & 1;    // 0..1

    const int Ktot  = Cin * 9;
    const int HoWo  = Ho * Wo;
    const int nBase = blockIdx.x * 64;    // pixel base within this batch image
    const int mBase = blockIdx.y * 64;
    const int bb    = blockIdx.z;         // batch index from grid -> no division

    const TIn* inb = in + (size_t)bb * Cin * Hin * Win;

    // Per-thread fill assignment: row = tid/4 (0..63), k-octet = (tid&3)*8
    const int frow = tid >> 2;
    const int koff = (tid & 3) * 8;

    // This thread's B-row (output pixel) coordinates (one /Wo division total).
    const int n_p   = nBase + frow;
    const bool nval = (n_p < HoWo);
    int poh = 0, pow_ = 0;
    if (nval) { poh = n_p / Wo; pow_ = n_p - poh * Wo; }
    const int ihB = poh * stride - pad;
    const int iwB = pow_ * stride - pad;
    const bool xform = (insc != nullptr);

    const _Float16* wrow = wf + (size_t)(mBase + frow) * Ktot;

    v8f c0 = {}; v8f c1 = {};

    for (int k0 = 0; k0 < Ktot; k0 += 32) {
        // ---- A tile: f16 weights [Cout][Cin*9], contiguous octet ----
        {
            v8h t;
            #pragma unroll
            for (int j = 0; j < 8; ++j) {
                int k = k0 + koff + j;
                t[j] = (k < Ktot) ? wrow[k] : (_Float16)0.0f;
            }
            *(v8h*)&As[frow * 40 + koff] = t;
            // Prefetch next K-step's weight octet (global_prefetch_b8;
            // speculative -> silently dropped past the end of the tensor).
            __builtin_prefetch(wrow + k0 + 32 + koff, 0, 1);
        }
        // ---- B tile: im2col octet with fused BN+ReLU on read ----
        {
            v8h t;
            #pragma unroll
            for (int j = 0; j < 8; ++j) {
                int k = k0 + koff + j;
                float v = 0.0f;
                if (nval && k < Ktot) {
                    int cin = k / 9;          // literal divisor: mul/shift
                    int r   = k - cin * 9;
                    int kh  = r / 3;
                    int kw  = r - kh * 3;
                    int ih  = ihB + kh, iw = iwB + kw;
                    if ((unsigned)ih < (unsigned)Hin &&
                        (unsigned)iw < (unsigned)Win) {
                        v = (float)inb[((size_t)cin * Hin + ih) * Win + iw];
                        if (xform) {
                            v = v * insc[cin] + insh[cin];
                            v = v > 0.0f ? v : 0.0f;
                        }
                    }
                }
                t[j] = (_Float16)v;
            }
            *(v8h*)&Bs[frow * 48 + koff] = t;
        }
        __syncthreads();

        // ---- WMMA fragments (contiguous k = 16*(lane/16) + s, same order in A and B) ----
        // A 16x32: lane<16 -> M=lane, K=0..15; lane>=16 -> K=16..31
        const int am  = mi * 16 + (lane & 15);
        const int akb = (lane < 16) ? 0 : 16;
        v8h a0 = *(const v8h*)&As[am * 40 + akb];
        v8h a1 = *(const v8h*)&As[am * 40 + akb + 8];
        v16h afr = __builtin_shufflevector(a0, a1,
            0,1,2,3,4,5,6,7, 8,9,10,11,12,13,14,15);

        // B 32x16: lane<16 -> N=lane, K=0..15; lane>=16 -> K=16..31
        const int bn  = ni * 32 + (lane & 15);
        const int bkb = (lane < 16) ? 0 : 16;
        v16h bf0 = *(const v16h*)&Bs[bn * 48 + bkb];
        v16h bf1 = *(const v16h*)&Bs[(bn + 16) * 48 + bkb];

        c0 = __builtin_amdgcn_wmma_f32_16x16x32_f16(
                 false, afr, false, bf0, (short)0, c0, false, false);
        c1 = __builtin_amdgcn_wmma_f32_16x16x32_f16(
                 false, afr, false, bf1, (short)0, c1, false, false);
        __syncthreads();
    }

    // ---- Epilogue. D layout: lane holds column n=(lane&15), rows
    //      m = mi*16 + r + (lane<16 ? 0 : 8), r = 0..7 ----
    TOut* outb = out + (size_t)bb * Cout * HoWo;
    const float gv = (mode != 0) ? gate[bb] : 1.0f;
    const int mrowB = mBase + mi * 16 + ((lane < 16) ? 0 : 8);
    #pragma unroll
    for (int half = 0; half < 2; ++half) {
        int ncol = nBase + ni * 32 + half * 16 + (lane & 15);
        if (ncol >= HoWo) continue;
        const v8f& cc = half ? c1 : c0;
        #pragma unroll
        for (int r = 0; r < 8; ++r) {
            int m = mrowB + r;
            float val = cc[r] + bias[m];
            size_t oidx = (size_t)m * HoWo + ncol;
            if (mode == 0)       outb[oidx] = (TOut)val;
            else if (mode == 1)  outb[oidx] = (TOut)(gv * val);
            else                 outb[oidx] = (TOut)((float)outb[oidx] + gv * val);
        }
    }
}

// ---------------------------------------------------------------------------
// Per-channel batch-norm statistics over f16 activation [B,C,HW].
// One block per channel -> scale = g*rsqrt(var+eps), shift = b - mean*scale.
// ---------------------------------------------------------------------------
__global__ __launch_bounds__(256) void bn_stats_kernel(
    const _Float16* __restrict__ act, const float* __restrict__ g,
    const float* __restrict__ bta, float* __restrict__ sc,
    float* __restrict__ sh, int B, int C, int HW)
{
    const int c = blockIdx.x;
    const int tid = threadIdx.x;
    __shared__ float rs[256], rq[256];
    float s = 0.0f, q = 0.0f;
    for (int b = 0; b < B; ++b) {
        const _Float16* p = act + ((size_t)b * C + c) * HW;
        for (int i = tid; i < HW; i += 256) {
            float v = (float)p[i];
            s += v; q += v * v;
        }
    }
    rs[tid] = s; rq[tid] = q;
    __syncthreads();
    for (int o = 128; o > 0; o >>= 1) {
        if (tid < o) { rs[tid] += rs[tid + o]; rq[tid] += rq[tid + o]; }
        __syncthreads();
    }
    if (tid == 0) {
        float n = (float)B * (float)HW;
        float mean = rs[0] / n;
        float var  = rq[0] / n - mean * mean;
        float inv  = rsqrtf(var + 1e-5f);
        float scv  = g[c] * inv;
        sc[c] = scv;
        sh[c] = bta[c] - mean * scv;
    }
}

// ---------------------------------------------------------------------------
// relu + spatial mean over e5 output (B=8, C=1024, 900 px) -> pooled[b][c]
// ---------------------------------------------------------------------------
__global__ __launch_bounds__(128) void pool_kernel(
    const _Float16* __restrict__ act5, float* __restrict__ pooled)
{
    const int c = blockIdx.x, b = blockIdx.y, tid = threadIdx.x;
    const _Float16* p = act5 + ((size_t)b * 1024 + c) * 900;
    __shared__ float red[128];
    float s = 0.0f;
    for (int i = tid; i < 900; i += 128) {
        float v = (float)p[i];
        s += v > 0.0f ? v : 0.0f;
    }
    red[tid] = s;
    __syncthreads();
    for (int o = 64; o > 0; o >>= 1) {
        if (tid < o) red[tid] += red[tid + o];
        __syncthreads();
    }
    if (tid == 0) pooled[b * 1024 + c] = red[0] * (1.0f / 900.0f);
}

// ---------------------------------------------------------------------------
// lat = relu(pooled @ pw.T + pb)  : (8,1024)x(256,1024) -> (8,256)
// ---------------------------------------------------------------------------
__global__ __launch_bounds__(256) void lat_kernel(
    const float* __restrict__ pooled, const float* __restrict__ pw,
    const float* __restrict__ pbv, float* __restrict__ lat)
{
    const int b = blockIdx.x, d = threadIdx.x;
    const float* pr = pooled + b * 1024;
    const float* wr = pw + (size_t)d * 1024;
    float acc = pbv[d];
    for (int k = 0; k < 1024; ++k) acc += pr[k] * wr[k];
    lat[b * 256 + d] = acc > 0.0f ? acc : 0.0f;
}

// ---------------------------------------------------------------------------
// Heads: gates(7), wb(3), gamma(1), omega(1) per batch from lat[b][256].
// params layout (floats): [0..55] gatesT[k*8+b], [56..79] wb[b*3+c],
//                         [80..87] gamma[b], [88..95] omega[b],
//                         [96..119] a[b*3+c] (filled by atm_kernel)
// ---------------------------------------------------------------------------
__global__ __launch_bounds__(256) void heads_kernel(
    const float* __restrict__ lat,
    const float* __restrict__ gw, const float* __restrict__ gb,
    const float* __restrict__ wbw, const float* __restrict__ wbb,
    const float* __restrict__ gaw, const float* __restrict__ gab,
    const float* __restrict__ dfw, const float* __restrict__ dfb,
    float* __restrict__ params)
{
    const int b = blockIdx.x, tid = threadIdx.x;
    __shared__ float L[256], red[256], outv[12];
    L[tid] = lat[b * 256 + tid];
    __syncthreads();
    for (int j = 0; j < 12; ++j) {
        const float* wrow; float bv;
        if (j < 7)       { wrow = gw  + j * 256;        bv = gb[j]; }
        else if (j < 10) { wrow = wbw + (j - 7) * 256;  bv = wbb[j - 7]; }
        else if (j == 10){ wrow = gaw;                  bv = gab[0]; }
        else             { wrow = dfw;                  bv = dfb[0]; }
        red[tid] = L[tid] * wrow[tid];
        __syncthreads();
        for (int o = 128; o > 0; o >>= 1) {
            if (tid < o) red[tid] += red[tid + o];
            __syncthreads();
        }
        if (tid == 0) outv[j] = red[0] + bv;
        __syncthreads();
    }
    if (tid == 0) {
        for (int k = 0; k < 7; ++k)
            params[k * 8 + b] = trf(outv[k], 0.01f, 1.0f);
        float w0 = expf(trf(outv[7], -0.5f, 0.5f));
        float w1 = expf(trf(outv[8], -0.5f, 0.5f));
        float w2 = expf(trf(outv[9], -0.5f, 0.5f));
        float cs = 1.0f / (1e-5f + 0.27f * w0 + 0.67f * w1 + 0.06f * w2);
        params[56 + b * 3 + 0] = cs * w0;
        params[56 + b * 3 + 1] = cs * w1;
        params[56 + b * 3 + 2] = cs * w2;
        float lg = logf(2.5f);
        params[80 + b] = expf(trf(outv[10], -lg, lg));
        params[88 + b] = trf(outv[11], 0.1f, 1.0f);
    }
}

// ---------------------------------------------------------------------------
// Atmospheric light: per batch, top ~numpx brightest dark-channel pixels'
// mean color (histogram-threshold selection over dark in [0,1]).
// ---------------------------------------------------------------------------
__global__ __launch_bounds__(256) void atm_kernel(
    const float* __restrict__ x, float* __restrict__ params, int HWs)
{
    const int b = blockIdx.x, tid = threadIdx.x;
    const int numpx = max(HWs / 1000, 1);
    __shared__ unsigned hist[1024];
    __shared__ int sT;
    __shared__ float ssum[4];
    for (int i = tid; i < 1024; i += 256) hist[i] = 0u;
    if (tid < 4) ssum[tid] = 0.0f;
    __syncthreads();
    const float* x0 = x + ((size_t)b * 3 + 0) * HWs;
    const float* x1 = x + ((size_t)b * 3 + 1) * HWs;
    const float* x2 = x + ((size_t)b * 3 + 2) * HWs;
    for (int i = tid; i < HWs; i += 256) {
        float d = fminf(x0[i], fminf(x1[i], x2[i]));
        int bin = (int)(d * 1024.0f);
        bin = bin < 0 ? 0 : (bin > 1023 ? 1023 : bin);
        atomicAdd(&hist[bin], 1u);
    }
    __syncthreads();
    if (tid == 0) {
        int acc = 0, T = 0;
        for (int i = 1023; i >= 0; --i) {
            acc += (int)hist[i];
            if (acc >= numpx) { T = i; break; }
        }
        sT = T;
    }
    __syncthreads();
    float thresh = (float)sT * (1.0f / 1024.0f);
    for (int i = tid; i < HWs; i += 256) {
        float v0 = x0[i], v1 = x1[i], v2 = x2[i];
        float d = fminf(v0, fminf(v1, v2));
        if (d >= thresh) {
            atomicAdd(&ssum[0], v0);
            atomicAdd(&ssum[1], v1);
            atomicAdd(&ssum[2], v2);
            atomicAdd(&ssum[3], 1.0f);
        }
    }
    __syncthreads();
    if (tid < 3) {
        float cnt = fmaxf(ssum[3], 1.0f);
        params[96 + b * 3 + tid] = ssum[tid] / cnt;
    }
}

// ---------------------------------------------------------------------------
// Init global min/max keys (mins -> 0xFFFFFFFF, maxes -> 0).
// ---------------------------------------------------------------------------
__global__ void init_mm_kernel(unsigned* __restrict__ mm) {
    int i = threadIdx.x;
    if (i < 6) mm[i] = (i & 1) ? 0u : 0xFFFFFFFFu;
}

// ---------------------------------------------------------------------------
// Enhancement math shared by the two passes.
// ---------------------------------------------------------------------------
__device__ __forceinline__ void enhance_vals(
    const float* __restrict__ x, const float* __restrict__ params,
    int b, int p, int HWs, float wbv[3], float gmv[3], float dfv[3])
{
    float xc[3];
    #pragma unroll
    for (int c = 0; c < 3; ++c)
        xc[c] = x[((size_t)b * 3 + c) * HWs + p];
    float ga = params[80 + b];
    float om = params[88 + b];
    float a0 = params[96 + b * 3 + 0];
    float a1 = params[96 + b * 3 + 1];
    float a2 = params[96 + b * 3 + 2];
    float i = fminf(xc[0] / a0, fminf(xc[1] / a1, xc[2] / a2));
    float t = fmaxf(1.0f - om * i, 0.01f);
    float av[3] = {a0, a1, a2};
    #pragma unroll
    for (int c = 0; c < 3; ++c) {
        wbv[c] = params[56 + b * 3 + c] * xc[c];
        gmv[c] = powf(fmaxf(xc[c], 1e-4f), ga);
        dfv[c] = (xc[c] - av[c]) / t + av[c];
    }
}

// Pass 1: global min/max of each enhanced image (ordered-uint atomics).
__global__ __launch_bounds__(256) void enhance_minmax_kernel(
    const float* __restrict__ x, const float* __restrict__ params,
    unsigned* __restrict__ mm, int N, int HWs)
{
    const int idx = blockIdx.x * 256 + threadIdx.x;
    const int tid = threadIdx.x;
    float lmn[3] = {3.4e38f, 3.4e38f, 3.4e38f};
    float lmx[3] = {-3.4e38f, -3.4e38f, -3.4e38f};
    if (idx < N) {
        int b = idx / HWs, p = idx - b * HWs;
        float wv[3], gv[3], dv[3];
        enhance_vals(x, params, b, p, HWs, wv, gv, dv);
        #pragma unroll
        for (int c = 0; c < 3; ++c) {
            lmn[0] = fminf(lmn[0], wv[c]); lmx[0] = fmaxf(lmx[0], wv[c]);
            lmn[1] = fminf(lmn[1], gv[c]); lmx[1] = fmaxf(lmx[1], gv[c]);
            lmn[2] = fminf(lmn[2], dv[c]); lmx[2] = fmaxf(lmx[2], dv[c]);
        }
    }
    __shared__ float red[256];
    for (int cat = 0; cat < 3; ++cat) {
        red[tid] = lmn[cat];
        __syncthreads();
        for (int o = 128; o > 0; o >>= 1) {
            if (tid < o) red[tid] = fminf(red[tid], red[tid + o]);
            __syncthreads();
        }
        if (tid == 0) atomicMin(&mm[cat * 2 + 0], fkey(red[0]));
        __syncthreads();
        red[tid] = lmx[cat];
        __syncthreads();
        for (int o = 128; o > 0; o >>= 1) {
            if (tid < o) red[tid] = fmaxf(red[tid], red[tid + o]);
            __syncthreads();
        }
        if (tid == 0) atomicMax(&mm[cat * 2 + 1], fkey(red[0]));
        __syncthreads();
    }
}

// Pass 2: recompute + normalize to [0,1], write f16 enhanced images.
__global__ __launch_bounds__(256) void enhance_write_kernel(
    const float* __restrict__ x, const float* __restrict__ params,
    const unsigned* __restrict__ mm,
    _Float16* __restrict__ wbimg, _Float16* __restrict__ gmimg,
    _Float16* __restrict__ dfimg, int N, int HWs)
{
    const int idx = blockIdx.x * 256 + threadIdx.x;
    if (idx >= N) return;
    int b = idx / HWs, p = idx - b * HWs;
    float wv[3], gv[3], dv[3];
    enhance_vals(x, params, b, p, HWs, wv, gv, dv);
    float mn0 = funkey(mm[0]), mx0 = funkey(mm[1]);
    float mn1 = funkey(mm[2]), mx1 = funkey(mm[3]);
    float mn2 = funkey(mm[4]), mx2 = funkey(mm[5]);
    float r0 = 1.0f / fmaxf(mx0 - mn0, 1e-12f);
    float r1 = 1.0f / fmaxf(mx1 - mn1, 1e-12f);
    float r2 = 1.0f / fmaxf(mx2 - mn2, 1e-12f);
    #pragma unroll
    for (int c = 0; c < 3; ++c) {
        size_t o = ((size_t)b * 3 + c) * HWs + p;
        wbimg[o] = (_Float16)((wv[c] - mn0) * r0);
        gmimg[o] = (_Float16)((gv[c] - mn1) * r1);
        dfimg[o] = (_Float16)((dv[c] - mn2) * r2);
    }
}

// ---------------------------------------------------------------------------
// Host-side orchestration
// ---------------------------------------------------------------------------
extern "C" void kernel_launch(void* const* d_in, const int* in_sizes, int n_in,
                              void* d_out, int out_size, void* d_ws, size_t ws_size,
                              hipStream_t stream) {
    (void)in_sizes; (void)n_in; (void)out_size; (void)ws_size;
    const int B = 8;
    const int HWs = 512 * 512;

    const float* x    = (const float*)d_in[0];
    const float* e1w  = (const float*)d_in[1];  const float* e1b  = (const float*)d_in[2];
    const float* bn1g = (const float*)d_in[3];  const float* bn1b = (const float*)d_in[4];
    const float* e2w  = (const float*)d_in[5];  const float* e2b  = (const float*)d_in[6];
    const float* bn2g = (const float*)d_in[7];  const float* bn2b = (const float*)d_in[8];
    const float* e3w  = (const float*)d_in[9];  const float* e3b  = (const float*)d_in[10];
    const float* bn3g = (const float*)d_in[11]; const float* bn3b = (const float*)d_in[12];
    const float* e4w  = (const float*)d_in[13]; const float* e4b  = (const float*)d_in[14];
    const float* bn4g = (const float*)d_in[15]; const float* bn4b = (const float*)d_in[16];
    const float* e5w  = (const float*)d_in[17]; const float* e5b  = (const float*)d_in[18];
    const float* pw   = (const float*)d_in[19]; const float* pb   = (const float*)d_in[20];
    const float* gw   = (const float*)d_in[21]; const float* gb   = (const float*)d_in[22];
    const float* wbw  = (const float*)d_in[23]; const float* wbb  = (const float*)d_in[24];
    const float* gaw  = (const float*)d_in[25]; const float* gab  = (const float*)d_in[26];
    const float* dfw  = (const float*)d_in[27]; const float* dfb  = (const float*)d_in[28];
    const float* s0w  = (const float*)d_in[29]; const float* s0b  = (const float*)d_in[30];
    const float* s1w  = (const float*)d_in[31]; const float* s1b  = (const float*)d_in[32];
    const float* s2w  = (const float*)d_in[33]; const float* s2b  = (const float*)d_in[34];
    const float* mng  = (const float*)d_in[35]; const float* mnb  = (const float*)d_in[36];
    const float* mcw  = (const float*)d_in[37]; const float* mcb  = (const float*)d_in[38];

    // Workspace carve-up (all regions fully rewritten each launch).
    char* base = (char*)d_ws;
    size_t off = 0;
    auto take = [&](size_t bytes) -> char* {
        char* p = base + off;
        off += (bytes + 255) & ~(size_t)255;
        return p;
    };
    const size_t A1 = (size_t)8 * 64 * 65536;   // 256x256
    const size_t A2 = (size_t)8 * 128 * 16384;  // 128x128
    const size_t A3 = (size_t)8 * 256 * 4096;   // 64x64
    const size_t A4 = (size_t)8 * 512 * 1024;   // 32x32
    const size_t A5 = (size_t)8 * 1024 * 900;   // 30x30
    const size_t IMG = (size_t)8 * 3 * HWs;

    _Float16* act1  = (_Float16*)take(A1 * 2);
    _Float16* act2  = (_Float16*)take(A2 * 2);
    _Float16* act3  = (_Float16*)take(A3 * 2);
    _Float16* act4  = (_Float16*)take(A4 * 2);
    _Float16* act5  = (_Float16*)take(A5 * 2);
    _Float16* wbimg = (_Float16*)take(IMG * 2);
    _Float16* gmimg = (_Float16*)take(IMG * 2);
    _Float16* dfimg = (_Float16*)take(IMG * 2);
    _Float16* fused = (_Float16*)take(A1 * 2);
    float* pooled = (float*)take(8 * 1024 * 4);
    float* lat    = (float*)take(8 * 256 * 4);
    float* sc1 = (float*)take(64 * 4);  float* sh1 = (float*)take(64 * 4);
    float* sc2 = (float*)take(128 * 4); float* sh2 = (float*)take(128 * 4);
    float* sc3 = (float*)take(256 * 4); float* sh3 = (float*)take(256 * 4);
    float* sc4 = (float*)take(512 * 4); float* sh4 = (float*)take(512 * 4);
    float* scM = (float*)take(64 * 4);  float* shM = (float*)take(64 * 4);
    float* params = (float*)take(128 * 4);
    unsigned* mm  = (unsigned*)take(6 * 4);

    // f16 weight copies
    const int W1 = 64 * 27, W2 = 128 * 576, W3 = 256 * 1152, W4 = 512 * 2304;
    const int W5 = 1024 * 4608, WS = 64 * 27, WM = 64 * 576;
    _Float16* e1wh = (_Float16*)take((size_t)W1 * 2);
    _Float16* e2wh = (_Float16*)take((size_t)W2 * 2);
    _Float16* e3wh = (_Float16*)take((size_t)W3 * 2);
    _Float16* e4wh = (_Float16*)take((size_t)W4 * 2);
    _Float16* e5wh = (_Float16*)take((size_t)W5 * 2);
    _Float16* s0wh = (_Float16*)take((size_t)WS * 2);
    _Float16* s1wh = (_Float16*)take((size_t)WS * 2);
    _Float16* s2wh = (_Float16*)take((size_t)WS * 2);
    _Float16* mcwh = (_Float16*)take((size_t)WM * 2);

    const dim3 blk(256);
    auto wgrid = [](int n) { return dim3((n + 255) / 256); };
    convert_w_kernel<<<wgrid(W1), blk, 0, stream>>>(e1w, e1wh, W1);
    convert_w_kernel<<<wgrid(W2), blk, 0, stream>>>(e2w, e2wh, W2);
    convert_w_kernel<<<wgrid(W3), blk, 0, stream>>>(e3w, e3wh, W3);
    convert_w_kernel<<<wgrid(W4), blk, 0, stream>>>(e4w, e4wh, W4);
    convert_w_kernel<<<wgrid(W5), blk, 0, stream>>>(e5w, e5wh, W5);
    convert_w_kernel<<<wgrid(WS), blk, 0, stream>>>(s0w, s0wh, WS);
    convert_w_kernel<<<wgrid(WS), blk, 0, stream>>>(s1w, s1wh, WS);
    convert_w_kernel<<<wgrid(WS), blk, 0, stream>>>(s2w, s2wh, WS);
    convert_w_kernel<<<wgrid(WM), blk, 0, stream>>>(mcw, mcwh, WM);

    // grid: x = pixel tiles within one batch image, y = Cout tiles, z = batch
    auto cgrid = [B](int HoWo, int Cout) {
        return dim3((HoWo + 63) / 64, Cout / 64, B);
    };

    // ---- Encoder ----
    conv_wmma_kernel<float, _Float16><<<cgrid(65536, 64), blk, 0, stream>>>(
        x, e1wh, e1b, act1, nullptr, nullptr, nullptr, 0,
        3, 512, 512, 64, 256, 256, 2, 1);
    bn_stats_kernel<<<64, blk, 0, stream>>>(act1, bn1g, bn1b, sc1, sh1, B, 64, 65536);

    conv_wmma_kernel<_Float16, _Float16><<<cgrid(16384, 128), blk, 0, stream>>>(
        act1, e2wh, e2b, act2, sc1, sh1, nullptr, 0,
        64, 256, 256, 128, 128, 128, 2, 1);
    bn_stats_kernel<<<128, blk, 0, stream>>>(act2, bn2g, bn2b, sc2, sh2, B, 128, 16384);

    conv_wmma_kernel<_Float16, _Float16><<<cgrid(4096, 256), blk, 0, stream>>>(
        act2, e3wh, e3b, act3, sc2, sh2, nullptr, 0,
        128, 128, 128, 256, 64, 64, 2, 1);
    bn_stats_kernel<<<256, blk, 0, stream>>>(act3, bn3g, bn3b, sc3, sh3, B, 256, 4096);

    conv_wmma_kernel<_Float16, _Float16><<<cgrid(1024, 512), blk, 0, stream>>>(
        act3, e4wh, e4b, act4, sc3, sh3, nullptr, 0,
        256, 64, 64, 512, 32, 32, 2, 1);
    bn_stats_kernel<<<512, blk, 0, stream>>>(act4, bn4g, bn4b, sc4, sh4, B, 512, 1024);

    // e5: stride 1, pad 0, bn4+relu fused on read; relu applied in pool
    conv_wmma_kernel<_Float16, _Float16><<<cgrid(900, 1024), blk, 0, stream>>>(
        act4, e5wh, e5b, act5, sc4, sh4, nullptr, 0,
        512, 32, 32, 1024, 30, 30, 1, 0);

    // ---- Heads ----
    pool_kernel<<<dim3(1024, 8), dim3(128), 0, stream>>>(act5, pooled);
    lat_kernel<<<8, blk, 0, stream>>>(pooled, pw, pb, lat);
    heads_kernel<<<8, blk, 0, stream>>>(lat, gw, gb, wbw, wbb, gaw, gab, dfw, dfb, params);
    atm_kernel<<<8, blk, 0, stream>>>(x, params, HWs);

    // ---- Enhanced images (global norm01 via ordered-uint atomics) ----
    const int Npix = B * HWs;
    init_mm_kernel<<<1, 32, 0, stream>>>(mm);
    enhance_minmax_kernel<<<(Npix + 255) / 256, blk, 0, stream>>>(x, params, mm, Npix, HWs);
    enhance_write_kernel<<<(Npix + 255) / 256, blk, 0, stream>>>(
        x, params, mm, wbimg, gmimg, dfimg, Npix, HWs);

    // ---- Gated stem convs fused into one f16 buffer ----
    conv_wmma_kernel<_Float16, _Float16><<<cgrid(65536, 64), blk, 0, stream>>>(
        wbimg, s0wh, s0b, fused, nullptr, nullptr, params + 0 * 8, 1,
        3, 512, 512, 64, 256, 256, 2, 1);
    conv_wmma_kernel<_Float16, _Float16><<<cgrid(65536, 64), blk, 0, stream>>>(
        gmimg, s1wh, s1b, fused, nullptr, nullptr, params + 1 * 8, 2,
        3, 512, 512, 64, 256, 256, 2, 1);
    conv_wmma_kernel<_Float16, _Float16><<<cgrid(65536, 64), blk, 0, stream>>>(
        dfimg, s2wh, s2b, fused, nullptr, nullptr, params + 2 * 8, 2,
        3, 512, 512, 64, 256, 256, 2, 1);

    // ---- Final: bn(fused)+relu fused into read side of mc conv -> d_out f32
    bn_stats_kernel<<<64, blk, 0, stream>>>(fused, mng, mnb, scM, shM, B, 64, 65536);
    conv_wmma_kernel<_Float16, float><<<cgrid(65536, 64), blk, 0, stream>>>(
        fused, mcwh, mcb, (float*)d_out, scM, shM, nullptr, 0,
        64, 256, 256, 64, 256, 256, 1, 1);
}
